// ShallowEncoderRepat_63170378989795
// MI455X (gfx1250) — matrix operand order repaired and checked by reading, not compiled, not run on hardware
//
#include <hip/hip_runtime.h>
#include <math.h>

typedef _Float16 v8h  __attribute__((ext_vector_type(8)));
typedef _Float16 v16h __attribute__((ext_vector_type(16)));
typedef float    v8f  __attribute__((ext_vector_type(8)));
typedef unsigned int v4u __attribute__((ext_vector_type(4)));
typedef int      v4i  __attribute__((ext_vector_type(4)));
typedef int      v8i  __attribute__((ext_vector_type(8)));

#define BB   16
#define HH   128
#define WW   128
#define C1   16
#define C2   64
#define HWSZ (HH*WW)
#define TT   4
#define NPER (BB*HWSZ)          // elements per channel for BN stats (262144)

// ---- workspace layout (float offsets) ----
#define W2H_OFF 0               // repacked conv2 weights: 5*64*32 halfs = 5120 floats
#define SUM1    5120
#define SSQ1    5136
#define SUM2    5152
#define SSQ2    5216
#define A1OFF   5280
#define B1OFF   5296
#define A2OFF   5312
#define B2OFF   5376
#define DECOFF  5440            // decay = 1 - alpha*(1-tau)
#define BIASOFF 5504            // (1-alpha)*lin
#define OMGOFF  5568            // (1-gamma)
#define SUBOFF  5632
#define VTHOFF  5696
#define INPCOFF 5760            // 4*64: 1 - beta*(1-conduct[t])
#define H1_OFF  6144            // B*C1*HW floats (conv1 raw)
#define H2_OFF  (H1_OFF + BB*C1*HWSZ)

__device__ __forceinline__ float sigm(float x) { return 1.0f / (1.0f + expf(-x)); }

// ---------------- zero the BN-stat accumulators ----------------
__global__ void k_init(float* ws) {
    int t = threadIdx.x;
    if (t < 160) ws[SUM1 + t] = 0.0f;
}

// ---------------- conv1 (3->16) + BN1 batch-stat accumulation ----------------
__global__ void k_conv1(const float* __restrict__ x, const float* __restrict__ w1,
                        float* __restrict__ ws) {
    __shared__ float wsm[16 * 3 * 9];
    __shared__ float s_sum[16], s_ssq[16];
    const int b = blockIdx.x >> 7, y = blockIdx.x & 127;
    const int tid = threadIdx.x;
    for (int i = tid; i < 432; i += 128) wsm[i] = w1[i];
    if (tid < 16) { s_sum[tid] = 0.0f; s_ssq[tid] = 0.0f; }
    __syncthreads();

    float in[3][3][3];
    #pragma unroll
    for (int ci = 0; ci < 3; ++ci)
      #pragma unroll
      for (int dy = 0; dy < 3; ++dy)
        #pragma unroll
        for (int dx = 0; dx < 3; ++dx) {
            int ys = y + dy - 1, xs = tid + dx - 1;
            float v = 0.0f;
            if (ys >= 0 && ys < HH && xs >= 0 && xs < WW)
                v = x[((b * 3 + ci) * HH + ys) * WW + xs];
            in[ci][dy][dx] = v;
        }

    #pragma unroll
    for (int co = 0; co < 16; ++co) {
        float acc = 0.0f;
        #pragma unroll
        for (int ci = 0; ci < 3; ++ci)
          #pragma unroll
          for (int dy = 0; dy < 3; ++dy)
            #pragma unroll
            for (int dx = 0; dx < 3; ++dx)
                acc = fmaf(wsm[(co * 3 + ci) * 9 + dy * 3 + dx], in[ci][dy][dx], acc);
        ws[H1_OFF + ((b * C1 + co) * HH + y) * WW + tid] = acc;
        atomicAdd(&s_sum[co], acc);
        atomicAdd(&s_ssq[co], acc * acc);
    }
    __syncthreads();
    if (tid < 16) {
        atomicAdd(&ws[SUM1 + tid], s_sum[tid]);
        atomicAdd(&ws[SSQ1 + tid], s_ssq[tid]);
    }
}

// ---------------- repack conv2 weights into f16 K-chunk layout ----------------
// w2h[kk][n][klocal]: chunk kk covers taps 2kk,2kk+1; klocal = tap_in_pair*16 + cin
__global__ void k_repack(const float* __restrict__ w2, float* __restrict__ ws) {
    int i = blockIdx.x * 256 + threadIdx.x;          // 5*2048 = 10240
    if (i >= 10240) return;
    int kk = i / 2048, r = i % 2048;
    int n = r >> 5, kl = r & 31;
    int tap = kk * 2 + (kl >> 4), c = kl & 15;
    float v = (tap < 9) ? w2[(n * 16 + c) * 9 + tap] : 0.0f;
    ((_Float16*)(ws + W2H_OFF))[i] = (_Float16)v;
}

// ---------------- finalize BN1 ----------------
__global__ void k_fin1(const float* __restrict__ bw, const float* __restrict__ bb,
                       float* __restrict__ ws) {
    int c = threadIdx.x;
    if (c >= 16) return;
    float mean = ws[SUM1 + c] / (float)NPER;
    float var  = ws[SSQ1 + c] / (float)NPER - mean * mean;
    float a = bw[c] * rsqrtf(var + 1e-5f);
    ws[A1OFF + c] = a;
    ws[B1OFF + c] = bb[c] - mean * a;
}

// ---------------- conv2 (16->64) implicit GEMM via WMMA + TDM B-load ----------------
// Block = one image row (128 px), 8 waves. Wave w: pixels [16w,16w+16) x 64 out-ch.
// K = 144 (16 cin x 9 taps) as 5 chunks of K=32 (2 taps each, last padded).
// A operand: LDS row cache [dy][x+1][cin] built once (BN1+ReLU fused, f16);
//            im2col is implicit in fragment addressing (2x ds_load_b128 / frag).
// B operand: all 5 chunks DMA'd by ONE tensor_load_to_lds with LDS padding
//            (16 dwords data + 4 dwords pad) into stride-40-half rows.
__global__ void k_conv2(float* __restrict__ ws) {
    __shared__ __align__(16) _Float16 RC[3 * 130 * 16];   // row cache, 12480 B
    __shared__ __align__(16) _Float16 Bl[5 * 64 * 40];    // padded B tiles, 25600 B
    __shared__ float s_sum[64], s_ssq[64], s_a1[16], s_b1[16];

    const int b = blockIdx.x >> 7, y = blockIdx.x & 127;
    const int tid = threadIdx.x, lane = tid & 31, wid = tid >> 5;
    const float* h1r = ws + H1_OFF;

    if (tid < 16) { s_a1[tid] = ws[A1OFF + tid]; s_b1[tid] = ws[B1OFF + tid]; }
    if (tid < 64) { s_sum[tid] = 0.0f; s_ssq[tid] = 0.0f; }

    // ---- issue TDM load of all B chunks (wave 0 only; EXEC-independent DMA) ----
    if (wid == 0) {
        unsigned long long gaddr = (unsigned long long)(ws + W2H_OFF);
        unsigned int laddr = (unsigned int)(unsigned long long)&Bl[0];   // LDS byte addr
        // D# group0: count=1 | lds_addr | global_addr[56:0] | type=2
        v4u g0 = { 1u, laddr, (unsigned int)gaddr,
                   (unsigned int)((gaddr >> 32) & 0x01FFFFFFull) | 0x80000000u };
        // D# group1: data_size=2B, pad_enable, pad_interval=3 (16 dwords),
        //            pad_amount=3 (4 dwords); 1-D tile of 10240 elements.
        v8i g1 = { (int)((1u << 16) | (1u << 20) | (3u << 22) | (3u << 25)),
                   (int)(10240u << 16),   // tensor_dim0 lo16 in [31:16]
                   0,
                   (int)(10240u << 16),   // tile_dim0 in [31:16]
                   0,
                   10240,                 // tensor_dim0_stride lo32
                   0, 0 };
        v4i g2 = { 0, 0, 0, 0 };
        v4i g3 = { 0, 0, 0, 0 };
#if __has_include(<hip/amd_detail/amd_gfx1250_TDM.h>)
        v8i g4 = { 0, 0, 0, 0, 0, 0, 0, 0 };
        __builtin_amdgcn_tensor_load_to_lds(g0, g1, g2, g3, g4, 0);
#else
        __builtin_amdgcn_tensor_load_to_lds(g0, g1, g2, g3, 0);
#endif
    }
    __syncthreads();   // s_a1/s_b1 visible before row-cache build

    // ---- build f16 row cache: RC[dy][xp][c], xp = x+1 with zero halo ----
    #pragma unroll
    for (int it = 0; it < 24; ++it) {
        int i = it * 256 + tid;                 // 6144 = 3 dy * 16 c * 128 x
        int xx = i & 127, c = (i >> 7) & 15, dy = i >> 11;
        int ys = y + dy - 1;
        float v = 0.0f;
        if (ys >= 0 && ys < HH) {
            float raw = h1r[((b * C1 + c) * HH + ys) * WW + xx];
            v = fmaxf(fmaf(raw, s_a1[c], s_b1[c]), 0.0f);   // BN1 + ReLU
        }
        RC[((dy * 130) + xx + 1) * 16 + c] = (_Float16)v;
    }
    if (tid < 96) {                             // zero halo columns xp=0,129
        int c = tid & 15, e = (tid >> 4) & 1, dy = tid / 32;
        RC[((dy * 130) + e * 129) * 16 + c] = (_Float16)0.0f;
    }
    if (wid == 0) __builtin_amdgcn_s_wait_tensorcnt(0);  // TDM done (wave 0)
    __syncthreads();                            // RC + Bl visible to all waves

    // ---- main K loop: pure fragment loads + WMMA, no staging, no barriers ----
    v8f acc[4];
    #pragma unroll
    for (int nt = 0; nt < 4; ++nt) acc[nt] = (v8f){0.f,0.f,0.f,0.f,0.f,0.f,0.f,0.f};

    const int m0 = wid * 16;
    const int px = m0 + (lane & 15);
    const int coff = (lane < 16) ? 0 : 8;   // A: lanes0-15 K{0..7,16..23}; 16-31 K{8..15,24..31}
    const int kb   = (lane < 16) ? 0 : 8;   // B: lanes0-15 K{0..7,16..23}; 16-31 K{8..15,24..31}

    #pragma unroll
    for (int kk = 0; kk < 5; ++kk) {
        const int t0 = kk * 2, t1 = kk * 2 + 1;
        v8h alo = *(const v8h*)&RC[((t0 / 3) * 130 + px + (t0 % 3)) * 16 + coff];
        v8h ahi = (t1 < 9)
            ? *(const v8h*)&RC[((t1 / 3) * 130 + px + (t1 % 3)) * 16 + coff]
            : (v8h){(_Float16)0,(_Float16)0,(_Float16)0,(_Float16)0,
                    (_Float16)0,(_Float16)0,(_Float16)0,(_Float16)0};
        v16h af = __builtin_shufflevector(alo, ahi, 0,1,2,3,4,5,6,7,8,9,10,11,12,13,14,15);
        #pragma unroll
        for (int nt = 0; nt < 4; ++nt) {
            int nrow = (kk * 64 + nt * 16 + (lane & 15)) * 40 + kb;
            v8h blo = *(const v8h*)&Bl[nrow];
            v8h bhi = *(const v8h*)&Bl[nrow + 16];
            v16h bf = __builtin_shufflevector(blo, bhi, 0,1,2,3,4,5,6,7,8,9,10,11,12,13,14,15);
            acc[nt] = __builtin_amdgcn_wmma_f32_16x16x32_f16(
                false, af, false, bf, (short)0, acc[nt], false, false);
        }
    }

    // ---- write raw conv2 output + accumulate BN2 batch stats ----
    float* h2r = ws + H2_OFF;
    #pragma unroll
    for (int nt = 0; nt < 4; ++nt) {
        int ch = nt * 16 + (lane & 15);
        float ps = 0.0f, pq = 0.0f;
        #pragma unroll
        for (int r = 0; r < 8; ++r) {
            int xo = m0 + r + ((lane >> 4) << 3);    // D: VGPR r -> M = r + 8*(lane/16)
            float v = acc[nt][r];
            h2r[((b * C2 + ch) * HH + y) * WW + xo] = v;
            ps += v; pq += v * v;
        }
        atomicAdd(&s_sum[ch], ps);
        atomicAdd(&s_ssq[ch], pq);
    }
    __syncthreads();
    if (tid < 64) {
        atomicAdd(&ws[SUM2 + tid], s_sum[tid]);
        atomicAdd(&ws[SSQ2 + tid], s_ssq[tid]);
    }
}

// ---------------- finalize BN2 + GLIF channel-wise coefficients ----------------
__global__ void k_fin2(const float* __restrict__ bw, const float* __restrict__ bb,
                       const float* __restrict__ gate_raw, const float* __restrict__ tau_raw,
                       const float* __restrict__ vth_raw, const float* __restrict__ lin_raw,
                       const float* __restrict__ sub_raw, const float* __restrict__ cond_raw,
                       float* __restrict__ ws) {
    int c = threadIdx.x;
    if (c >= 64) return;
    float mean = ws[SUM2 + c] / (float)NPER;
    float var  = ws[SSQ2 + c] / (float)NPER - mean * mean;
    float a = bw[c] * rsqrtf(var + 1e-5f);
    ws[A2OFF + c] = a;
    ws[B2OFF + c] = bb[c] - mean * a;

    float alpha = sigm(gate_raw[c]);
    float beta  = sigm(gate_raw[64 + c]);
    float gamma = sigm(gate_raw[128 + c]);
    float tau   = sigm(tau_raw[c]);
    ws[DECOFF  + c] = 1.0f - alpha * (1.0f - tau);
    ws[BIASOFF + c] = (1.0f - alpha) * sigm(lin_raw[c]);
    ws[OMGOFF  + c] = 1.0f - gamma;
    ws[SUBOFF  + c] = sigm(sub_raw[c]);
    ws[VTHOFF  + c] = sigm(vth_raw[c]);
    #pragma unroll
    for (int t = 0; t < TT; ++t)
        ws[INPCOFF + t * 64 + c] = 1.0f - beta * (1.0f - sigm(cond_raw[t * 64 + c]));
}

// ---------------- fused BN2 + GLIF time loop, float4 streaming ----------------
__global__ void k_glif(const float* __restrict__ ws, float* __restrict__ out) {
    const int idx4 = blockIdx.x * 256 + threadIdx.x;
    const int c = (idx4 >> 12) & 63;
    const float* h2r = ws + H2_OFF;

    const float a2  = ws[A2OFF  + c], b2  = ws[B2OFF  + c];
    const float dec = ws[DECOFF + c], bia = ws[BIASOFF + c];
    const float omg = ws[OMGOFF + c], sub = ws[SUBOFF + c];
    const float vth = ws[VTHOFF + c];

    const float4 raw = *(const float4*)&h2r[idx4 * 4];
    float h[4] = { fmaf(raw.x, a2, b2), fmaf(raw.y, a2, b2),
                   fmaf(raw.z, a2, b2), fmaf(raw.w, a2, b2) };
    float v[4] = {0.f, 0.f, 0.f, 0.f};

    #pragma unroll
    for (int t = 0; t < TT; ++t) {
        const float ic = ws[INPCOFF + t * 64 + c];
        float sp[4];
        #pragma unroll
        for (int j = 0; j < 4; ++j) {
            float u = fmaf(dec, v[j], fmaf(ic, h[j], -bia));
            float s = (u >= vth) ? 1.0f : 0.0f;
            sp[j] = s;
            v[j] = (s != 0.0f) ? omg * (u - sub) : u;
        }
        float4 s4; s4.x = sp[0]; s4.y = sp[1]; s4.z = sp[2]; s4.w = sp[3];
        *(float4*)&out[t * (BB * C2 * HWSZ) + idx4 * 4] = s4;
    }
}

extern "C" void kernel_launch(void* const* d_in, const int* in_sizes, int n_in,
                              void* d_out, int out_size, void* d_ws, size_t ws_size,
                              hipStream_t stream) {
    const float* x        = (const float*)d_in[0];
    const float* w1       = (const float*)d_in[1];
    const float* bn1_w    = (const float*)d_in[2];
    const float* bn1_b    = (const float*)d_in[3];
    const float* w2       = (const float*)d_in[4];
    const float* bn2_w    = (const float*)d_in[5];
    const float* bn2_b    = (const float*)d_in[6];
    const float* gate_raw = (const float*)d_in[7];
    const float* tau_raw  = (const float*)d_in[8];
    const float* vth_raw  = (const float*)d_in[9];
    const float* lin_raw  = (const float*)d_in[10];
    const float* sub_raw  = (const float*)d_in[11];
    const float* cond_raw = (const float*)d_in[12];
    float* ws  = (float*)d_ws;
    float* out = (float*)d_out;

    k_init  <<<1, 256, 0, stream>>>(ws);
    k_conv1 <<<BB * HH, 128, 0, stream>>>(x, w1, ws);
    k_repack<<<40, 256, 0, stream>>>(w2, ws);
    k_fin1  <<<1, 32, 0, stream>>>(bn1_w, bn1_b, ws);
    k_conv2 <<<BB * HH, 256, 0, stream>>>(ws);
    k_fin2  <<<1, 64, 0, stream>>>(bn2_w, bn2_b, gate_raw, tau_raw, vth_raw,
                                   lin_raw, sub_raw, cond_raw, ws);
    k_glif  <<<(BB * C2 * HWSZ) / 4 / 256, 256, 0, stream>>>(ws, out);
}
